// ResidualGIN_53566832115779
// MI455X (gfx1250) — matrix operand order repaired and checked by reading, not compiled, not run on hardware
//
#include <hip/hip_runtime.h>
#include <cstdint>
#include <cstddef>

// Problem constants (match the reference).
#define NN 50000
#define EE 300000
#define DD 256
#define HH 256
#define CC 47
#define LL 6

typedef __bf16 bf16_t;
typedef __attribute__((ext_vector_type(16))) __bf16 v16bf;
typedef __attribute__((ext_vector_type(8)))  __bf16 v8bf;
typedef __attribute__((ext_vector_type(4)))  __bf16 v4bf;
typedef __attribute__((ext_vector_type(8)))  float  v8f;

// Low 32 bits of a generic pointer to LDS == LDS byte offset (ISA §10.2 aperture map).
__device__ __forceinline__ unsigned lds_off32(const void* p) {
    return (unsigned)(uintptr_t)p;
}
// Async global->LDS copy of 16 bytes (8 bf16), tracked by ASYNCcnt.
__device__ __forceinline__ void async_copy_b128(unsigned lds, const void* gptr) {
    asm volatile("global_load_async_to_lds_b128 %0, %1, off"
                 :: "v"(lds), "v"((uint64_t)(uintptr_t)gptr) : "memory");
}
__device__ __forceinline__ void wait_asynccnt0() {
    asm volatile("s_wait_asynccnt 0x0" ::: "memory");
}

// ---------------------------------------------------------------------------
// Weight prep: W (K x Ncols, f32 row-major) -> Wt (Ncols x K, bf16) once/launch
// ---------------------------------------------------------------------------
__global__ __launch_bounds__(256)
void k_wconv(const float* __restrict__ W, bf16_t* __restrict__ Wt, int K, int Ncols)
{
    int idx = blockIdx.x * 256 + threadIdx.x;   // over K*Ncols, coalesced reads
    if (idx >= K * Ncols) return;
    int k = idx / Ncols, c = idx % Ncols;
    Wt[(size_t)c * K + k] = (bf16_t)W[idx];
}

// ---------------------------------------------------------------------------
// GIN aggregation: agg = (1 + eps) * h  (eps read on device, graph-capture safe)
// ---------------------------------------------------------------------------
__global__ __launch_bounds__(256)
void k_agg_init(const float* __restrict__ h, float* __restrict__ agg,
                const float* __restrict__ eps, int li, int total4)
{
    int i = blockIdx.x * 256 + threadIdx.x;
    if (i >= total4) return;
    float s = 1.0f + eps[li];
    float4 v = ((const float4*)h)[i];
    v.x *= s; v.y *= s; v.z *= s; v.w *= s;
    ((float4*)agg)[i] = v;
}

// One wave per edge: agg[dst] += h[src] (256 floats, 8 per lane, L2 atomics).
__global__ __launch_bounds__(256)
void k_agg_edges(const float* __restrict__ h, const long long* __restrict__ ei,
                 float* __restrict__ agg)
{
    int e = blockIdx.x * 8 + (threadIdx.x >> 5);
    if (e >= EE) return;
    int lane = threadIdx.x & 31;
    int src = (int)ei[e];
    int dst = (int)ei[EE + e];
    const float* hs = h + (size_t)src * HH;
    float* ad = agg + (size_t)dst * HH;
#pragma unroll
    for (int i = 0; i < 8; ++i) {
        int c = lane + i * 32;
        atomicAdd(ad + c, hs[c]);
    }
}

// ---------------------------------------------------------------------------
// WMMA GEMM: Out = act(A @ W + bias)
//  - A: M x K row-major (f32 via VGPR-convert staging, or bf16 via async-LDS)
//  - Wt: Ncols x K row-major bf16 (pre-transposed) -> B tile async-copied to LDS
//  - 8 waves per block; each wave owns a 32x64 accumulator tile (2x4 frags)
//  - v_wmma_f32_16x16x32_bf16 with f32 accumulate
//  - Fragments are contiguous 16B LDS runs per the ISA VGPR layouts (§7.12.2)
// ---------------------------------------------------------------------------
template<int WM, int WN, bool RELU, bool ABF16, bool OUTBF16>
__global__ __launch_bounds__(32 * WM * WN)
void gemm_wmma(const void* __restrict__ Ap, const bf16_t* __restrict__ Wt,
               const float* __restrict__ bias, void* __restrict__ Outp,
               int M, int K, int Ncols, int ldOut)
{
    constexpr int MT  = WM * 32;       // block tile rows
    constexpr int NT  = WN * 64;       // block tile cols
    constexpr int KT  = 32;            // K per WMMA step
    constexpr int KP  = KT + 8;        // padded K stride: 40 bf16 = 80B (16B mult)
    constexpr int NTH = 32 * WM * WN;  // threads per block

    __shared__ alignas(16) bf16_t sA [MT][KP];   // [row][k]
    __shared__ alignas(16) bf16_t sBt[NT][KP];   // [col][k]

    const int tid  = threadIdx.x;
    const int lane = tid & 31;
    const int wave = tid >> 5;
    const int wm   = wave % WM;
    const int wn   = wave / WM;
    const int half = lane >> 4;    // 0: lanes 0-15, 1: lanes 16-31
    const int l16  = lane & 15;
    const int blockM = blockIdx.x * MT;
    const int blockN = blockIdx.y * NT;

    v8f acc[2][4];
#pragma unroll
    for (int i = 0; i < 2; ++i)
#pragma unroll
        for (int n = 0; n < 4; ++n)
#pragma unroll
            for (int r = 0; r < 8; ++r) acc[i][n][r] = 0.0f;

    for (int k0 = 0; k0 < K; k0 += KT) {
        // ---- stage A tile (MT x 32) ----
        if constexpr (ABF16) {
            // Async copy: each lane moves 8 contiguous bf16 (16B).
            for (int idx = tid; idx < MT * (KT / 8); idx += NTH) {
                int r  = idx & (MT - 1);
                int k8 = (idx / MT) * 8;
                int gr = blockM + r;
                if (gr < M)   // rows >= M: LDS left stale, never stored
                    async_copy_b128(lds_off32(&sA[r][k8]),
                                    (const bf16_t*)Ap + (size_t)gr * K + (k0 + k8));
            }
        } else {
            // f32 source: load float4, convert, ds_store 4 bf16.
            for (int base = tid * 4; base < MT * KT; base += NTH * 4) {
                int r  = base >> 5;
                int kk = base & 31;
                int gr = blockM + r;
                v4bf v;
                if (gr < M) {
                    const float4 f = *(const float4*)((const float*)Ap + (size_t)gr * K + (k0 + kk));
                    v[0] = (bf16_t)f.x; v[1] = (bf16_t)f.y;
                    v[2] = (bf16_t)f.z; v[3] = (bf16_t)f.w;
                    if (k0 + KT < K)
                        __builtin_prefetch((const char*)Ap +
                            ((size_t)gr * K + (k0 + KT + kk)) * 4, 0, 1);
                } else {
                    v[0] = v[1] = v[2] = v[3] = (bf16_t)0.0f;
                }
                *(v4bf*)&sA[r][kk] = v;
            }
        }
        // ---- stage B tile: sBt[c][k] <= Wt[blockN+c][k0+k], async 16B/lane ----
        for (int idx = tid; idx < NT * (KT / 8); idx += NTH) {
            int c  = idx & (NT - 1);
            int k8 = (idx / NT) * 8;
            int gc = blockN + c;
            if (gc < Ncols)   // cols >= Ncols: LDS stale, never stored
                async_copy_b128(lds_off32(&sBt[c][k8]),
                                Wt + (size_t)gc * K + (k0 + k8));
        }
        wait_asynccnt0();
        __syncthreads();

        // ---- build fragments (contiguous 16B LDS loads) and issue WMMAs ----
        v16bf afr[2], bfr[4];
#pragma unroll
        for (int i = 0; i < 2; ++i) {
            const int row = wm * 32 + i * 16 + l16;
            const v8bf* ar = (const v8bf*)&sA[row][0];
            // lanes 0-15: K 0-7 then 16-23 ; lanes 16-31: K 8-15 then 24-31
            v8bf lo = ar[half];
            v8bf hi = ar[2 + half];
            afr[i] = __builtin_shufflevector(lo, hi,
                0, 1, 2, 3, 4, 5, 6, 7, 8, 9, 10, 11, 12, 13, 14, 15);
        }
#pragma unroll
        for (int n = 0; n < 4; ++n) {
            const int col = wn * 64 + n * 16 + l16;
            const v8bf* br = (const v8bf*)&sBt[col][0];
            // lanes 0-15: K 0-7 then 16-23 ; lanes 16-31: K 8-15 then 24-31
            v8bf lo = br[half];
            v8bf hi = br[2 + half];
            bfr[n] = __builtin_shufflevector(lo, hi,
                0, 1, 2, 3, 4, 5, 6, 7, 8, 9, 10, 11, 12, 13, 14, 15);
        }
#pragma unroll
        for (int i = 0; i < 2; ++i)
#pragma unroll
            for (int n = 0; n < 4; ++n)
                acc[i][n] = __builtin_amdgcn_wmma_f32_16x16x32_bf16(
                    false, afr[i], false, bfr[n], (short)0, acc[i][n], false, false);
        __syncthreads();
    }

    // ---- epilogue: bias (+ReLU), store f32 or bf16 ----
    // C/D layout: lane 0-15 -> M=r, lane 16-31 -> M=8+r ; N = lane&15.
#pragma unroll
    for (int i = 0; i < 2; ++i)
#pragma unroll
        for (int n = 0; n < 4; ++n) {
            const int col = blockN + wn * 64 + n * 16 + l16;
            if (col < Ncols) {
                const float bb = bias[col];
#pragma unroll
                for (int r = 0; r < 8; ++r) {
                    const int row = blockM + wm * 32 + i * 16 + half * 8 + r;
                    if (row < M) {
                        float v = acc[i][n][r] + bb;
                        if constexpr (RELU) v = fmaxf(v, 0.0f);
                        if constexpr (OUTBF16)
                            ((bf16_t*)Outp)[(size_t)row * ldOut + col] = (bf16_t)v;
                        else
                            ((float*)Outp)[(size_t)row * ldOut + col] = v;
                    }
                }
            }
        }
}

// ---------------------------------------------------------------------------
// BatchNorm statistics + apply
// ---------------------------------------------------------------------------
__global__ __launch_bounds__(512)
void k_zero512(float* p) { p[threadIdx.x] = 0.0f; }

__global__ __launch_bounds__(256)
void k_stats(const float* __restrict__ Z, float* __restrict__ sums,
             float* __restrict__ sumsq)
{
    int col = threadIdx.x;            // 256 threads == 256 channels, coalesced rows
    int r0 = blockIdx.x * 256;
    int rend = min(r0 + 256, NN);
    float s = 0.0f, q = 0.0f;
    for (int r = r0; r < rend; ++r) {
        float v = Z[(size_t)r * HH + col];
        s += v; q += v * v;
    }
    atomicAdd(&sums[col], s);
    atomicAdd(&sumsq[col], q);
}

__global__ __launch_bounds__(256)
void k_stats_fin(const float* __restrict__ sums, const float* __restrict__ sumsq,
                 const float* __restrict__ gamma, const float* __restrict__ beta,
                 float* __restrict__ ab)
{
    int c = threadIdx.x;
    float mu  = sums[c]  * (1.0f / NN);
    float var = sumsq[c] * (1.0f / NN) - mu * mu;   // biased var, matches jnp.var
    float inv = rsqrtf(var + 1e-5f);
    float a = gamma[c] * inv;
    ab[c]      = a;
    ab[HH + c] = beta[c] - mu * a;
}

__global__ __launch_bounds__(256)
void k_bn_apply(const float* __restrict__ Z, const float* __restrict__ ab,
                float* __restrict__ h, int hasres, int total4)
{
    int i = blockIdx.x * 256 + threadIdx.x;
    if (i >= total4) return;
    int cb = (i * 4) & (HH - 1);
    float4 z = ((const float4*)Z)[i];
    float4 o;
    o.x = fmaxf(z.x * ab[cb + 0] + ab[HH + cb + 0], 0.0f);
    o.y = fmaxf(z.y * ab[cb + 1] + ab[HH + cb + 1], 0.0f);
    o.z = fmaxf(z.z * ab[cb + 2] + ab[HH + cb + 2], 0.0f);
    o.w = fmaxf(z.w * ab[cb + 3] + ab[HH + cb + 3], 0.0f);
    if (hasres) {
        float4 r = ((const float4*)h)[i];
        o.x += r.x; o.y += r.y; o.z += r.z; o.w += r.w;
    }
    ((float4*)h)[i] = o;
}

// ---------------------------------------------------------------------------
// Orchestration
// ---------------------------------------------------------------------------
extern "C" void kernel_launch(void* const* d_in, const int* in_sizes, int n_in,
                              void* d_out, int out_size, void* d_ws, size_t ws_size,
                              hipStream_t stream)
{
    const float*     x     = (const float*)d_in[0];
    const long long* ei    = (const long long*)d_in[1];
    const float*     W1s   = (const float*)d_in[2];
    const float*     b1s   = (const float*)d_in[3];
    const float*     W2s   = (const float*)d_in[4];
    const float*     b2s   = (const float*)d_in[5];
    const float*     eps   = (const float*)d_in[6];
    const float*     gamma = (const float*)d_in[7];
    const float*     beta  = (const float*)d_in[8];
    const float*     Wl1   = (const float*)d_in[9];
    const float*     bl1   = (const float*)d_in[10];
    const float*     Wl2   = (const float*)d_in[11];
    const float*     bl2   = (const float*)d_in[12];
    float*           out   = (float*)d_out;

    // Workspace: agg/Z (f32 N*H) | h (f32 N*H) | Y (bf16 N*H) | stats 4KB | Wt bf16
    char* ws = (char*)d_ws;
    const size_t nh = (size_t)NN * HH;
    float*  agg   = (float*)ws;
    float*  h     = (float*)(ws + nh * 4);
    bf16_t* ybf   = (bf16_t*)(ws + 2 * nh * 4);
    float*  stats = (float*)(ws + 2 * nh * 4 + nh * 2);
    float*  sums  = stats;
    float*  sumsq = stats + HH;
    float*  ab    = stats + 2 * HH;
    float*  Z     = agg;   // agg is dead after GEMM1; reuse for Z

    bf16_t* wt    = (bf16_t*)(ws + 2 * nh * 4 + nh * 2 + 4096);
    const size_t wm1 = (size_t)DD * HH;          // 65536 per matrix
    bf16_t* W1t  = wt;                           // 5 x [H][D]
    bf16_t* W2t  = wt + 5 * wm1;                 // 5 x [H][H]
    bf16_t* Wl1t = wt + 10 * wm1;                // [H][H]
    bf16_t* Wl2t = wt + 11 * wm1;                // [C][H]

    const int total4 = (NN * HH) / 4;
    dim3 b256(256);
    dim3 gElem((total4 + 255) / 256);
    dim3 gEdges((EE + 7) / 8);
    dim3 gGemm((NN + 127) / 128, HH / 128);
    dim3 gGemmF((NN + 255) / 256, 1);
    dim3 gStats((NN + 255) / 256);
    dim3 gW((DD * HH + 255) / 256);
    dim3 gWl2((HH * CC + 255) / 256);

    // One-time (per launch) weight convert+transpose to bf16.
    for (int i = 0; i < LL - 1; ++i) {
        k_wconv<<<gW, b256, 0, stream>>>(W1s + (size_t)i * wm1, W1t + (size_t)i * wm1, DD, HH);
        k_wconv<<<gW, b256, 0, stream>>>(W2s + (size_t)i * wm1, W2t + (size_t)i * wm1, HH, HH);
    }
    k_wconv<<<gW,   b256, 0, stream>>>(Wl1, Wl1t, HH, HH);
    k_wconv<<<gWl2, b256, 0, stream>>>(Wl2, Wl2t, HH, CC);

    for (int i = 0; i < LL - 1; ++i) {
        const float* hin = (i == 0) ? x : h;
        k_agg_init<<<gElem, b256, 0, stream>>>(hin, agg, eps, i, total4);
        k_agg_edges<<<gEdges, b256, 0, stream>>>(hin, ei, agg);
        // Y = relu(agg @ W1 + b1), bf16 out
        gemm_wmma<4, 2, true, false, true><<<gGemm, b256, 0, stream>>>(
            agg, W1t + (size_t)i * wm1, b1s + (size_t)i * HH, ybf, NN, DD, HH, HH);
        // Z = Y @ W2 + b2, f32 out
        gemm_wmma<4, 2, false, true, false><<<gGemm, b256, 0, stream>>>(
            ybf, W2t + (size_t)i * wm1, b2s + (size_t)i * HH, Z, NN, HH, HH, HH);
        k_zero512<<<1, 512, 0, stream>>>(stats);
        k_stats<<<gStats, b256, 0, stream>>>(Z, sums, sumsq);
        k_stats_fin<<<1, b256, 0, stream>>>(sums, sumsq, gamma + (size_t)i * HH,
                                            beta + (size_t)i * HH, ab);
        k_bn_apply<<<gElem, b256, 0, stream>>>(Z, ab, h, (i > 0) ? 1 : 0, total4);
    }
    // Final GINConv: agg -> relu(agg@Wl1+bl1) -> @Wl2+bl2 (47 cols)
    k_agg_init<<<gElem, b256, 0, stream>>>(h, agg, eps, LL - 1, total4);
    k_agg_edges<<<gEdges, b256, 0, stream>>>(h, ei, agg);
    gemm_wmma<4, 2, true, false, true><<<gGemm, b256, 0, stream>>>(
        agg, Wl1t, bl1, ybf, NN, HH, HH, HH);
    gemm_wmma<8, 1, false, true, false><<<gGemmF, b256, 0, stream>>>(
        ybf, Wl2t, bl2, out, NN, HH, CC, CC);
}
